// MHAttention_38122129719421
// MI455X (gfx1250) — matrix operand order repaired and checked by reading, not compiled, not run on hardware
//
#include <hip/hip_runtime.h>

typedef __bf16 bf16;
typedef bf16 v16bf __attribute__((ext_vector_type(16)));
typedef bf16 v8bf  __attribute__((ext_vector_type(8)));
typedef bf16 v4bf  __attribute__((ext_vector_type(4)));
typedef float v8f  __attribute__((ext_vector_type(8)));

#define DEVINL __device__ __forceinline__

// ---- CDNA5 WMMA fragment helpers (wave32) -------------------------------
// A fragment (16x32 bf16, 8 VGPRs/lane): lane l holds row m=l%16, half=l/16.
//   elems e<8 : K = half*8 + e        (contiguous 8 -> one b128 LDS read)
//   elems e>=8: K = 16 + half*8 + e-8 (contiguous 8 -> one b128 LDS read)
DEVINL v16bf fragA(const bf16* row, int half) {
    v8bf lo = *(const v8bf*)(row + half * 8);
    v8bf hi = *(const v8bf*)(row + 16 + half * 8);
    return __builtin_shufflevector(lo, hi, 0,1,2,3,4,5,6,7,8,9,10,11,12,13,14,15);
}
// B fragment (32x16 bf16): lane l holds col n=l%16, half=l/16, K same as A:
// e<8: K = half*8 + e; e>=8: K = 16 + half*8 + e-8. Row pointer is B^T row (n-major).
DEVINL v16bf fragB(const bf16* row, int half) {
    v8bf lo = *(const v8bf*)(row + half * 8);
    v8bf hi = *(const v8bf*)(row + 16 + half * 8);
    return __builtin_shufflevector(lo, hi, 0,1,2,3,4,5,6,7,8,9,10,11,12,13,14,15);
}
DEVINL v8f wmma_bf16(v16bf a, v16bf b, v8f c) {
    return __builtin_amdgcn_wmma_f32_16x16x32_bf16(
        /*neg_a=*/false, a, /*neg_b=*/false, b,
        /*c_mod=*/(short)0, c, /*reuse_a=*/false, /*reuse_b=*/false);
}

// ---- Problem constants ---------------------------------------------------
constexpr int QN = 32, KN = 64, BS = 2048, H = 8, D = 64, E = 512;

// =========================================================================
// Batched per-slot GEMM: C[slot] = A[slot] (MxK) * W[slot] (KxN)
// A: AT (fp32 or bf16), W: fp32, C: OT (bf16 or fp32). M=2048,K=512,N=512.
// Block tile 128x256, 256 threads = 8 waves (2x4), wave tile 64x64 (4x4 WMMA).
// LDS double-buffered, one barrier per K-chunk: staging of chunk k+1 overlaps
// the 16 WMMAs of chunk k (stageA/stageB issued after the WMMA block).
// =========================================================================
template <typename AT, typename OT>
__global__ __launch_bounds__(256) void gemm_slot(const AT* __restrict__ A,
                                                 const float* __restrict__ W,
                                                 OT* __restrict__ C,
                                                 int M, int K, int N) {
    constexpr int BM = 128, BN = 256, BK = 32, LP = BK + 8;  // pad: conflict-free
    __shared__ __align__(16) bf16 sA[2][BM][LP];   // [buf][m][k]
    __shared__ __align__(16) bf16 sB[2][BN][LP];   // [buf][n][k]  (W transposed)

    const int slot = blockIdx.z;
    const AT*    Ab = A + (size_t)slot * M * K;
    const float* Wb = W + (size_t)slot * K * N;
    OT*          Cb = C + (size_t)slot * M * N;

    const int m0 = blockIdx.x * BM, n0 = blockIdx.y * BN;
    const int t = threadIdx.x;
    const int wid = t >> 5, lane = t & 31, half = lane >> 4, l16 = lane & 15;
    const int wr = wid & 1, wc = wid >> 1;   // wave tile origin: (wr*64, wc*64)

    // ---- staging helpers ------------------------------------------------
    auto stageA = [&](int buf, int kc) {      // 128x32, 16 elems/thread
        const int rb = t >> 3, c4 = (t & 7) << 2;
#pragma unroll
        for (int u = 0; u < 4; ++u) {
            const int r = rb + 32 * u;
            const AT* src = Ab + (size_t)(m0 + r) * K + kc + c4;
            bf16* dst = &sA[buf][r][c4];
            if constexpr (sizeof(AT) == 4) {
                const float4 v = *(const float4*)src;
                dst[0] = (bf16)v.x; dst[1] = (bf16)v.y;
                dst[2] = (bf16)v.z; dst[3] = (bf16)v.w;
            } else {
                *(v4bf*)dst = *(const v4bf*)src;
            }
        }
    };
    auto stageB = [&](int buf, int kc) {      // 32x256 -> transposed, 32 el/thread
        const int k = t & 31, nb = t >> 5;
#pragma unroll
        for (int u = 0; u < 8; ++u) {
            const int n = (nb + 8 * u) << 2;
            const float4 v = *(const float4*)(Wb + (size_t)(kc + k) * N + n0 + n);
            sB[buf][n + 0][k] = (bf16)v.x; sB[buf][n + 1][k] = (bf16)v.y;
            sB[buf][n + 2][k] = (bf16)v.z; sB[buf][n + 3][k] = (bf16)v.w;
        }
    };

    const v8f vzero = {0.f,0.f,0.f,0.f,0.f,0.f,0.f,0.f};
    v8f acc[4][4];
#pragma unroll
    for (int i = 0; i < 4; ++i)
#pragma unroll
        for (int j = 0; j < 4; ++j) acc[i][j] = vzero;

    stageA(0, 0);
    stageB(0, 0);

    int cur = 0;
    for (int kc = 0; kc < K; kc += BK) {
        __syncthreads();                       // buf[cur] ready, buf[cur^1] free
        v16bf af[4], bfm[4];
#pragma unroll
        for (int mi = 0; mi < 4; ++mi)
            af[mi] = fragA(&sA[cur][wr * 64 + mi * 16 + l16][0], half);
#pragma unroll
        for (int ni = 0; ni < 4; ++ni)
            bfm[ni] = fragB(&sB[cur][wc * 64 + ni * 16 + l16][0], half);
#pragma unroll
        for (int mi = 0; mi < 4; ++mi)
#pragma unroll
            for (int ni = 0; ni < 4; ++ni)
                acc[mi][ni] = wmma_bf16(af[mi], bfm[ni], acc[mi][ni]);
        if (kc + BK < K) {                     // overlap: stage next chunk
            stageA(cur ^ 1, kc + BK);
            stageB(cur ^ 1, kc + BK);
        }
        cur ^= 1;
    }

    // ---- store D tiles: elem(m,n): vgpr g -> m = base + half*8 + g, n = base + l16
#pragma unroll
    for (int mi = 0; mi < 4; ++mi)
#pragma unroll
        for (int ni = 0; ni < 4; ++ni) {
            const int r0 = m0 + wr * 64 + mi * 16 + half * 8;
            const int c  = n0 + wc * 64 + ni * 16 + l16;
#pragma unroll
            for (int g = 0; g < 8; ++g)
                Cb[(size_t)(r0 + g) * N + c] = (OT)acc[mi][ni][g];
        }
}

// =========================================================================
// Fused attention: one block per batch b; 8 waves = 8 heads.
// logits = Q·K^T / 8, softmax over 64 keys, out = P·V.  All WMMA bf16.
// =========================================================================
__global__ __launch_bounds__(256) void attn_kernel(const bf16* __restrict__ Q,
                                                   const bf16* __restrict__ Ke,
                                                   const bf16* __restrict__ V,
                                                   bf16* __restrict__ O) {
    __shared__ __align__(16) bf16 sQ[QN][E + 8];       // [i][e]
    __shared__ __align__(16) bf16 sK[KN][E + 8];       // [j][e]
    __shared__ __align__(16) bf16 sVt[H][D][KN + 8];   // [h][d][j]  (V transposed)
    __shared__ __align__(16) bf16 sP[H][QN][KN + 8];   // [h][i][j]  softmax probs

    const int b = blockIdx.x;
    const int t = threadIdx.x;

    // ---- stage Q rows (32 x 512 bf16): thread -> row t/8, 64-col chunk ----
    {
        const int r = t >> 3, c0 = (t & 7) * 64;
        const bf16* src = Q + (size_t)r * (BS * E) + (size_t)b * E + c0;
#pragma unroll
        for (int u = 0; u < 8; ++u)
            *(v8bf*)&sQ[r][c0 + u * 8] = *(const v8bf*)(src + u * 8);
    }
    // ---- stage K rows (64 x 512) ----
    {
        const int r = t >> 2, c0 = (t & 3) * 128;
        const bf16* src = Ke + (size_t)r * (BS * E) + (size_t)b * E + c0;
#pragma unroll
        for (int u = 0; u < 16; ++u)
            *(v8bf*)&sK[r][c0 + u * 8] = *(const v8bf*)(src + u * 8);
    }
    // ---- stage V transposed per head: V[j][h*64+d] -> sVt[h][d][j] ----
    {
        const int j = t >> 2, e0 = (t & 3) * 128;
        const bf16* src = V + (size_t)j * (BS * E) + (size_t)b * E + e0;
#pragma unroll
        for (int u = 0; u < 16; ++u) {
            const v8bf v = *(const v8bf*)(src + u * 8);
            const int e = e0 + u * 8, h = e >> 6, d = e & 63;
#pragma unroll
            for (int x = 0; x < 8; ++x) sVt[h][d + x][j] = v[x];
        }
    }
    __syncthreads();

    const int wid = t >> 5, lane = t & 31, half = lane >> 4, l16 = lane & 15;
    const int h = wid;                       // one head per wave
    const v8f vzero = {0.f,0.f,0.f,0.f,0.f,0.f,0.f,0.f};

    // ---- logits (32 x 64) = Q_h (32x64) * K_h^T (64x64) ----
    v8f lg[2][4];
#pragma unroll
    for (int i = 0; i < 2; ++i)
#pragma unroll
        for (int j = 0; j < 4; ++j) lg[i][j] = vzero;

#pragma unroll
    for (int kb = 0; kb < D; kb += 32) {
        v16bf af[2], bfm[4];
#pragma unroll
        for (int mi = 0; mi < 2; ++mi)
            af[mi] = fragA(&sQ[mi * 16 + l16][h * D + kb], half);
#pragma unroll
        for (int ni = 0; ni < 4; ++ni)          // B[k=d][n=j]: K^T row j = key row
            bfm[ni] = fragB(&sK[ni * 16 + l16][h * D + kb], half);
#pragma unroll
        for (int mi = 0; mi < 2; ++mi)
#pragma unroll
            for (int ni = 0; ni < 4; ++ni)
                lg[mi][ni] = wmma_bf16(af[mi], bfm[ni], lg[mi][ni]);
    }

    // ---- row softmax over 64 keys (scale 1/sqrt(64)=0.125) ----
    const float invT = 0.125f;
#pragma unroll
    for (int mi = 0; mi < 2; ++mi)
#pragma unroll
        for (int g = 0; g < 8; ++g) {
            float m = lg[mi][0][g];
#pragma unroll
            for (int ni = 1; ni < 4; ++ni) m = fmaxf(m, lg[mi][ni][g]);
#pragma unroll
            for (int msk = 1; msk < 16; msk <<= 1)   // within 16-lane half
                m = fmaxf(m, __shfl_xor(m, msk, 32));
            float s = 0.f;
            float e[4];
#pragma unroll
            for (int ni = 0; ni < 4; ++ni) {
                e[ni] = __expf((lg[mi][ni][g] - m) * invT);
                s += e[ni];
            }
#pragma unroll
            for (int msk = 1; msk < 16; msk <<= 1)
                s += __shfl_xor(s, msk, 32);
            const float inv = 1.f / s;
            const int r = mi * 16 + half * 8 + g;
#pragma unroll
            for (int ni = 0; ni < 4; ++ni)
                sP[h][r][ni * 16 + l16] = (bf16)(e[ni] * inv);
        }
    __syncthreads();

    // ---- out (32 x 64) = P (32x64) * V_h (64x64) ----
    v8f oc[2][4];
#pragma unroll
    for (int i = 0; i < 2; ++i)
#pragma unroll
        for (int j = 0; j < 4; ++j) oc[i][j] = vzero;

#pragma unroll
    for (int kb = 0; kb < KN; kb += 32) {
        v16bf af[2], bfm[4];
#pragma unroll
        for (int mi = 0; mi < 2; ++mi)
            af[mi] = fragA(&sP[h][mi * 16 + l16][kb], half);
#pragma unroll
        for (int ni = 0; ni < 4; ++ni)          // B[k=j][n=d] = sVt[h][d][j]
            bfm[ni] = fragB(&sVt[h][ni * 16 + l16][kb], half);
#pragma unroll
        for (int mi = 0; mi < 2; ++mi)
#pragma unroll
            for (int ni = 0; ni < 4; ++ni)
                oc[mi][ni] = wmma_bf16(af[mi], bfm[ni], oc[mi][ni]);
    }

    // ---- store O[i][b][h*64+d] (bf16) ----
#pragma unroll
    for (int mi = 0; mi < 2; ++mi)
#pragma unroll
        for (int ni = 0; ni < 4; ++ni) {
            const int i0 = mi * 16 + half * 8;
            const int d  = ni * 16 + l16;
#pragma unroll
            for (int g = 0; g < 8; ++g)
                O[(size_t)(i0 + g) * (BS * E) + (size_t)b * E + h * D + d] =
                    (bf16)oc[mi][ni][g];
        }
}

// =========================================================================
extern "C" void kernel_launch(void* const* d_in, const int* in_sizes, int n_in,
                              void* d_out, int out_size, void* d_ws, size_t ws_size,
                              hipStream_t stream) {
    const float* q  = (const float*)d_in[0];   // (32, 2048, 512)
    const float* k  = (const float*)d_in[1];   // (64, 2048, 512)
    const float* qw = (const float*)d_in[2];   // (32, 512, 512)
    const float* kw = (const float*)d_in[3];   // (64, 512, 512)
    const float* vw = (const float*)d_in[4];   // (64, 512, 512)
    const float* pw = (const float*)d_in[5];   // (32, 512, 512)
    float* out = (float*)d_out;                // (32, 2048, 512)

    // workspace: bf16 intermediates
    bf16* qproj = (bf16*)d_ws;                          // 32*2048*512
    bf16* kproj = qproj + (size_t)QN * BS * E;          // 64*2048*512
    bf16* vproj = kproj + (size_t)KN * BS * E;          // 64*2048*512
    bf16* aout  = vproj + (size_t)KN * BS * E;          // 32*2048*512

    const dim3 blk(256);
    const dim3 gq(BS / 128, E / 256, QN);   // (16,2,32)
    const dim3 gk(BS / 128, E / 256, KN);   // (16,2,64)

    gemm_slot<float, bf16><<<gq, blk, 0, stream>>>(q, qw, qproj, BS, E, E);
    gemm_slot<float, bf16><<<gk, blk, 0, stream>>>(k, kw, kproj, BS, E, E);
    gemm_slot<float, bf16><<<gk, blk, 0, stream>>>(k, vw, vproj, BS, E, E);

    attn_kernel<<<dim3(BS), blk, 0, stream>>>(qproj, kproj, vproj, aout);

    gemm_slot<bf16, float><<<gq, blk, 0, stream>>>(aout, pw, out, BS, E, E);
}
